// Transformer_75144747811338
// MI455X (gfx1250) — matrix operand order repaired and checked by reading, not compiled, not run on hardware
//
#include <hip/hip_runtime.h>
#include <cstdint>

#define DEVFN __device__ __forceinline__

typedef __attribute__((ext_vector_type(16))) __bf16 v16bf;
typedef __attribute__((ext_vector_type(8)))  float  v8f;

static constexpr int Bc    = 8;
static constexpr int Nseq  = 1024;
static constexpr int Dm    = 256;
static constexpr int Hh    = 8;
static constexpr int NDEPTH = 4;
static constexpr int MLPD  = 1024;
static constexpr int INNER = Dm * Hh;     // 2048 (quirk: inner = dim * heads)
static constexpr int Mrows = Bc * Nseq;   // 8192

union FragU { v16bf v; uint4 u[2]; };

DEVFN __bf16 f2bf(float f) {
  union { float f; uint32_t u; } a; a.f = f;
  uint32_t r = a.u + 0x7FFFu + ((a.u >> 16) & 1u);   // round-to-nearest-even
  union { unsigned short s; __bf16 b; } o; o.s = (unsigned short)(r >> 16);
  return o.b;
}

DEVFN v8f vzero() {
  v8f v;
#pragma unroll
  for (int i = 0; i < 8; ++i) v[i] = 0.f;
  return v;
}

// A fragment, 16x32 bf16, row-major A with leading dim lda.
// ISA layout: lane l holds row m = l&15; VGPR0-3 => K in [8*half, 8*half+8),
// VGPR4-7 => K in [16+8*half, 16+8*half+8), half = l>>4.
DEVFN v16bf load_a_frag(const __bf16* __restrict__ A, int lda, int m, int k, int lane) {
  const int row  = m + (lane & 15);
  const int half = lane >> 4;
  const __bf16* p = A + (size_t)row * lda + k + 8 * half;
  FragU f;
  f.u[0] = *(const uint4*)(p);        // K = k+8*half .. +8
  f.u[1] = *(const uint4*)(p + 16);   // K = k+16+8*half .. +8
  return f.v;
}

// B fragment, 32x16 bf16, B stored COLUMN-major (Bt row-major [Ncols][K], ld ldb).
// ISA layout: lane l holds col n = l&15; VGPR0-3 => K in [8*half, 8*half+8), VGPR4-7 => K in [16+8*half, 16+8*half+8).
DEVFN v16bf load_b_frag(const __bf16* __restrict__ Bt, int ldb, int n, int k, int lane) {
  const int col  = n + (lane & 15);
  const int half = lane >> 4;
  const __bf16* p = Bt + (size_t)col * ldb + k + 8 * half;
  FragU f;
  f.u[0] = *(const uint4*)(p);
  f.u[1] = *(const uint4*)(p + 16);
  return f.v;
}

DEVFN v8f wmma_bf16(v16bf a, v16bf b, v8f c) {
  return __builtin_amdgcn_wmma_f32_16x16x32_bf16(false, a, false, b, (short)0, c, false, false);
}

// ---------------------------------------------------------------------------
// Weight prep: fp32 [L][R][C] -> bf16 [L][C][R]  (column-major B for WMMA)
// ---------------------------------------------------------------------------
__global__ __launch_bounds__(256) void transpose_to_bf16(
    const float* __restrict__ in, __bf16* __restrict__ out, int R, int C, int total) {
  int idx = blockIdx.x * 256 + threadIdx.x;
  if (idx >= total) return;
  int rc    = R * C;
  int layer = idx / rc;
  int rem   = idx - layer * rc;
  int r = rem / C, c = rem - r * C;
  out[(size_t)layer * rc + (size_t)c * R + r] = f2bf(in[idx]);
}

// ---------------------------------------------------------------------------
// LayerNorm over axis=1 (sequence): stats over n per (b,d) column; gamma/beta[n]
// One wave32 per column; output bf16 (GEMM A operand).
// ---------------------------------------------------------------------------
__global__ __launch_bounds__(256) void layernorm_axis1(
    const float* __restrict__ x, const float* __restrict__ g,
    const float* __restrict__ be, __bf16* __restrict__ out) {
  const int col  = blockIdx.x * 8 + (threadIdx.x >> 5);   // (b,d) flat in [0, B*D)
  const int lane = threadIdx.x & 31;
  const int b = col >> 8, d = col & 255;
  const float* xp = x + (size_t)b * Nseq * Dm + d;
  float s = 0.f, s2 = 0.f;
  for (int n = lane; n < Nseq; n += 32) {
    float v = xp[(size_t)n * Dm];
    s += v; s2 += v * v;
  }
#pragma unroll
  for (int o = 16; o; o >>= 1) { s += __shfl_xor(s, o); s2 += __shfl_xor(s2, o); }
  float mu = s * (1.f / Nseq);
  float var = s2 * (1.f / Nseq) - mu * mu;
  float rs = rsqrtf(var + 1e-3f);
  for (int n = lane; n < Nseq; n += 32) {
    float v = (xp[(size_t)n * Dm] - mu) * rs * g[n] + be[n];
    out[((size_t)b * Nseq + n) * Dm + d] = f2bf(v);
  }
}

// ---------------------------------------------------------------------------
// WMMA GEMM: C[M,N] = A[M,K](bf16,row-major) x B (bf16, column-major as Bt[N][K])
// Block 256 thr = 8 waves; block tile 128x128; wave tile 32x64 (2x4 frags).
// ---------------------------------------------------------------------------
enum { EPI_QKV = 0, EPI_BIAS_RES = 1, EPI_BIAS_GELU = 2 };

template <int EPI>
__global__ __launch_bounds__(256) void gemm_bf16_wmma(
    const __bf16* __restrict__ A, const __bf16* __restrict__ Bt,
    int M, int Ncols, int K,
    const float* __restrict__ bias, const float* __restrict__ resid,
    float* __restrict__ outF, __bf16* __restrict__ outBf,
    __bf16* __restrict__ outQ, __bf16* __restrict__ outK, __bf16* __restrict__ outVT) {
  (void)M;
  const int lane = threadIdx.x & 31;
  const int wave = threadIdx.x >> 5;
  const int wm = wave & 3, wn = wave >> 2;
  const int m0 = blockIdx.y * 128 + wm * 32;
  const int n0 = blockIdx.x * 128 + wn * 64;

  v8f acc[2][4];
#pragma unroll
  for (int i = 0; i < 2; ++i)
#pragma unroll
    for (int j = 0; j < 4; ++j) acc[i][j] = vzero();

  for (int kk = 0; kk < K; kk += 32) {
    if (kk + 32 < K) {  // CDNA5 global_prefetch of next K-slab
      __builtin_prefetch(A + (size_t)(m0 + (lane & 15)) * K + kk + 32, 0, 3);
      __builtin_prefetch(Bt + (size_t)(n0 + (lane & 15)) * K + kk + 32, 0, 3);
    }
    v16bf a0 = load_a_frag(A, K, m0, kk, lane);
    v16bf a1 = load_a_frag(A, K, m0 + 16, kk, lane);
    v16bf b0 = load_b_frag(Bt, K, n0, kk, lane);
    v16bf b1 = load_b_frag(Bt, K, n0 + 16, kk, lane);
    v16bf b2 = load_b_frag(Bt, K, n0 + 32, kk, lane);
    v16bf b3 = load_b_frag(Bt, K, n0 + 48, kk, lane);
    acc[0][0] = wmma_bf16(a0, b0, acc[0][0]);
    acc[0][1] = wmma_bf16(a0, b1, acc[0][1]);
    acc[0][2] = wmma_bf16(a0, b2, acc[0][2]);
    acc[0][3] = wmma_bf16(a0, b3, acc[0][3]);
    acc[1][0] = wmma_bf16(a1, b0, acc[1][0]);
    acc[1][1] = wmma_bf16(a1, b1, acc[1][1]);
    acc[1][2] = wmma_bf16(a1, b2, acc[1][2]);
    acc[1][3] = wmma_bf16(a1, b3, acc[1][3]);
  }

  const int half = lane >> 4, nl = lane & 15;
#pragma unroll
  for (int i = 0; i < 2; ++i)
#pragma unroll
    for (int j = 0; j < 4; ++j)
#pragma unroll
      for (int r = 0; r < 8; ++r) {
        const int row = m0 + 16 * i + r + 8 * half;   // C/D layout: M = r + 8*half
        const int col = n0 + 16 * j + nl;             // N = lane&15
        float val = acc[i][j][r];
        if (EPI == EPI_QKV) {
          const int which = col >> 11;                // 0:q 1:k 2:v
          const int within = col & 2047;
          const int hh = within >> 8, dd = within & 255;
          const int bb = row >> 10, nn = row & 1023;
          const size_t bh = (size_t)(bb * Hh + hh);
          __bf16 bv = f2bf(val);
          if (which == 0)      outQ[(bh * Nseq + nn) * Dm + dd] = bv;
          else if (which == 1) outK[(bh * Nseq + nn) * Dm + dd] = bv;
          else                 outVT[(bh * Dm + dd) * Nseq + nn] = bv;   // V transposed
        } else if (EPI == EPI_BIAS_RES) {
          const size_t idx = (size_t)row * Ncols + col;
          outF[idx] = resid[idx] + bias[col] + val;
        } else {  // EPI_BIAS_GELU (exact erf gelu), bf16 out
          float t = val + bias[col];
          float gg = 0.5f * t * (1.f + erff(t * 0.70710678118654752f));
          outBf[(size_t)row * Ncols + col] = f2bf(gg);
        }
      }
}

// ---------------------------------------------------------------------------
// Flash attention, per-head dim 256, scale 1/16.
// Block: 256 thr (8 waves) handles one (b,h) x 64 query rows; 16 KV tiles of 64.
// Waves: S-gemm -> (w&3)=row tile, (w>>2)=32-col half. PV: (w&3)=row tile,
// (w>>2)=128-dim half. O lives in registers (8 v8f frags / wave).
// ---------------------------------------------------------------------------
__global__ __launch_bounds__(256) void flash_attn_wmma(
    const __bf16* __restrict__ Q, const __bf16* __restrict__ Kmat,
    const __bf16* __restrict__ VT, __bf16* __restrict__ Z) {
  __shared__ __align__(16) float  Ss[64 * 64];
  __shared__ __align__(16) __bf16 Ps[64 * 64];
  __shared__ float mrow[64], lrow[64], arow[64];

  const int lane = threadIdx.x & 31;
  const int wave = threadIdx.x >> 5;
  const int mi = wave & 3;    // 16-row sub-tile
  const int di = wave >> 2;   // 128-dim half (PV) / 32-col half (S)
  const int half = lane >> 4, nl = lane & 15;

  const int bh = blockIdx.y;
  const int b = bh >> 3, h = bh & 7;
  const int q0 = blockIdx.x * 64;

  const __bf16* Qbh = Q    + (size_t)bh * Nseq * Dm;
  const __bf16* Kbh = Kmat + (size_t)bh * Nseq * Dm;
  const __bf16* Vbh = VT   + (size_t)bh * Dm * Nseq;   // [d][n]

  if (threadIdx.x < 64) { mrow[threadIdx.x] = -INFINITY; lrow[threadIdx.x] = 0.f; }
  v8f o[8];
#pragma unroll
  for (int j = 0; j < 8; ++j) o[j] = vzero();
  __syncthreads();

  for (int t = 0; t < Nseq / 64; ++t) {
    const int k0 = t * 64;
    // ---- S tile = (Q Kt) * scale : wave computes 16 rows x 32 cols ----
    v8f s0 = vzero(), s1 = vzero();
    for (int kk = 0; kk < Dm; kk += 32) {
      v16bf aq  = load_a_frag(Qbh, Dm, q0 + 16 * mi, kk, lane);
      v16bf bk0 = load_b_frag(Kbh, Dm, k0 + 32 * di, kk, lane);      // K row-major == col-major of Kt
      v16bf bk1 = load_b_frag(Kbh, Dm, k0 + 32 * di + 16, kk, lane);
      s0 = wmma_bf16(aq, bk0, s0);
      s1 = wmma_bf16(aq, bk1, s1);
    }
#pragma unroll
    for (int r = 0; r < 8; ++r) {
      Ss[(16 * mi + r + 8 * half) * 64 + 32 * di + nl]      = s0[r] * 0.0625f;
      Ss[(16 * mi + r + 8 * half) * 64 + 32 * di + 16 + nl] = s1[r] * 0.0625f;
    }
    __syncthreads();

    // ---- online softmax: one thread per query row ----
    if (threadIdx.x < 64) {
      const int r = threadIdx.x;
      float mOld = mrow[r];
      float mNew = mOld;
      for (int j = 0; j < 64; ++j) mNew = fmaxf(mNew, Ss[r * 64 + j]);
      float al = __expf(mOld - mNew);
      float ssum = 0.f;
      for (int j = 0; j < 64; ++j) {
        float p = __expf(Ss[r * 64 + j] - mNew);
        Ps[r * 64 + j] = f2bf(p);
        ssum += p;
      }
      lrow[r] = lrow[r] * al + ssum;
      mrow[r] = mNew;
      arow[r] = al;
    }
    __syncthreads();

    // ---- rescale O and accumulate P @ V (V stored transposed: col-major B) ----
    float alr[8];
#pragma unroll
    for (int r = 0; r < 8; ++r) alr[r] = arow[16 * mi + r + 8 * half];
#pragma unroll
    for (int j = 0; j < 8; ++j)
#pragma unroll
      for (int r = 0; r < 8; ++r) o[j][r] *= alr[r];
    for (int kk = 0; kk < 64; kk += 32) {
      v16bf ap = load_a_frag(&Ps[0], 64, 16 * mi, kk, lane);
#pragma unroll
      for (int j = 0; j < 8; ++j) {
        v16bf bv = load_b_frag(Vbh, Nseq, 128 * di + 16 * j, k0 + kk, lane);
        o[j] = wmma_bf16(ap, bv, o[j]);
      }
    }
    __syncthreads();
  }

  // ---- normalize by l and store z[b, n, h*256 + d] (bf16) ----
  float linv[8];
#pragma unroll
  for (int r = 0; r < 8; ++r) linv[r] = 1.f / lrow[16 * mi + r + 8 * half];
#pragma unroll
  for (int j = 0; j < 8; ++j)
#pragma unroll
    for (int r = 0; r < 8; ++r) {
      const int n = q0 + 16 * mi + r + 8 * half;
      const int d = 128 * di + 16 * j + nl;
      Z[((size_t)(b * Nseq + n)) * INNER + h * Dm + d] = f2bf(o[j][r] * linv[r]);
    }
}

// ---------------------------------------------------------------------------
// Host orchestration
// ---------------------------------------------------------------------------
extern "C" void kernel_launch(void* const* d_in, const int* in_sizes, int n_in,
                              void* d_out, int out_size, void* d_ws, size_t ws_size,
                              hipStream_t stream) {
  (void)in_sizes; (void)n_in; (void)out_size; (void)ws_size;
  const float* x    = (const float*)d_in[0];
  const float* ln1g = (const float*)d_in[1];
  const float* ln1b = (const float*)d_in[2];
  const float* wqkv = (const float*)d_in[3];
  const float* wout = (const float*)d_in[4];
  const float* bout = (const float*)d_in[5];
  const float* ln2g = (const float*)d_in[6];
  const float* ln2b = (const float*)d_in[7];
  const float* w1   = (const float*)d_in[8];
  const float* b1   = (const float*)d_in[9];
  const float* w2   = (const float*)d_in[10];
  const float* b2   = (const float*)d_in[11];

  char* ws = (char*)d_ws;
  size_t off = 0;
  auto wsalloc = [&](size_t bytes) -> void* {
    void* p = ws + off;
    off += (bytes + 255) & ~(size_t)255;
    return p;
  };
  float*  x_cur  = (float*)wsalloc((size_t)Mrows * Dm * 4);
  __bf16* h_bf   = (__bf16*)wsalloc((size_t)Mrows * Dm * 2);
  __bf16* q_bf   = (__bf16*)wsalloc((size_t)Mrows * INNER * 2);
  __bf16* k_bf   = (__bf16*)wsalloc((size_t)Mrows * INNER * 2);
  __bf16* vT_bf  = (__bf16*)wsalloc((size_t)Mrows * INNER * 2);
  __bf16* z_bf   = (__bf16*)wsalloc((size_t)Mrows * INNER * 2);
  __bf16* mlp_bf = (__bf16*)wsalloc((size_t)Mrows * MLPD * 2);
  __bf16* wqkvT  = (__bf16*)wsalloc((size_t)NDEPTH * Dm * 3 * INNER * 2);
  __bf16* woutT  = (__bf16*)wsalloc((size_t)NDEPTH * INNER * Dm * 2);
  __bf16* w1T    = (__bf16*)wsalloc((size_t)NDEPTH * Dm * MLPD * 2);
  __bf16* w2T    = (__bf16*)wsalloc((size_t)NDEPTH * MLPD * Dm * 2);

  hipMemcpyAsync(x_cur, x, (size_t)Mrows * Dm * 4, hipMemcpyDeviceToDevice, stream);

  // Weight prep: fp32 row-major -> bf16 column-major, all layers at once.
  {
    int tot = NDEPTH * Dm * 3 * INNER;
    transpose_to_bf16<<<(tot + 255) / 256, 256, 0, stream>>>(wqkv, wqkvT, Dm, 3 * INNER, tot);
    tot = NDEPTH * INNER * Dm;
    transpose_to_bf16<<<(tot + 255) / 256, 256, 0, stream>>>(wout, woutT, INNER, Dm, tot);
    tot = NDEPTH * Dm * MLPD;
    transpose_to_bf16<<<(tot + 255) / 256, 256, 0, stream>>>(w1, w1T, Dm, MLPD, tot);
    tot = NDEPTH * MLPD * Dm;
    transpose_to_bf16<<<(tot + 255) / 256, 256, 0, stream>>>(w2, w2T, MLPD, Dm, tot);
  }

  for (int i = 0; i < NDEPTH; ++i) {
    // PreNorm 1 -> bf16
    layernorm_axis1<<<(Bc * Dm) / 8, 256, 0, stream>>>(x_cur, ln1g + i * Nseq, ln1b + i * Nseq, h_bf);
    // QKV GEMM: 8192 x 6144 x 256, scatter into q / k / v^T
    gemm_bf16_wmma<EPI_QKV><<<dim3((3 * INNER) / 128, Mrows / 128), 256, 0, stream>>>(
        h_bf, wqkvT + (size_t)i * Dm * 3 * INNER, Mrows, 3 * INNER, Dm,
        nullptr, nullptr, nullptr, nullptr, q_bf, k_bf, vT_bf);
    // Flash attention -> z (bf16, [B,N,INNER])
    flash_attn_wmma<<<dim3(Nseq / 64, Bc * Hh), 256, 0, stream>>>(q_bf, k_bf, vT_bf, z_bf);
    // Out projection + bias + residual: x = x + z @ w_out + b_out
    gemm_bf16_wmma<EPI_BIAS_RES><<<dim3(Dm / 128, Mrows / 128), 256, 0, stream>>>(
        z_bf, woutT + (size_t)i * INNER * Dm, Mrows, Dm, INNER,
        bout + i * Dm, x_cur, x_cur, nullptr, nullptr, nullptr, nullptr);
    // PreNorm 2 -> bf16
    layernorm_axis1<<<(Bc * Dm) / 8, 256, 0, stream>>>(x_cur, ln2g + i * Nseq, ln2b + i * Nseq, h_bf);
    // MLP up + GELU(exact) -> bf16
    gemm_bf16_wmma<EPI_BIAS_GELU><<<dim3(MLPD / 128, Mrows / 128), 256, 0, stream>>>(
        h_bf, w1T + (size_t)i * Dm * MLPD, Mrows, MLPD, Dm,
        b1 + i * MLPD, nullptr, nullptr, mlp_bf, nullptr, nullptr, nullptr);
    // MLP down + bias + residual: x = x + gelu @ w2 + b2
    gemm_bf16_wmma<EPI_BIAS_RES><<<dim3(Dm / 128, Mrows / 128), 256, 0, stream>>>(
        mlp_bf, w2T + (size_t)i * MLPD * Dm, Mrows, Dm, MLPD,
        b2 + i * Dm, x_cur, x_cur, nullptr, nullptr, nullptr, nullptr);
  }

  hipMemcpyAsync(d_out, x_cur, (size_t)Mrows * Dm * 4, hipMemcpyDeviceToDevice, stream);
}
